// CriticNetwork_8031588844055
// MI455X (gfx1250) — matrix operand order repaired and checked by reading, not compiled, not run on hardware
//
#include <hip/hip_runtime.h>
#include <hip/hip_bf16.h>
#include <math.h>

typedef __attribute__((ext_vector_type(16))) _Float16 v16h;
typedef __attribute__((ext_vector_type(8)))  float    v8f;

// ---------------------------------------------------------------------------
// WMMA GEMM: Y[M,N] = act(X[M,K] @ W[K,N] + bias)  (+ optional sigmoid-skip
// residual blend:  Y = sk*o + (1-sk)*R ). f16 via LDS staged in FRAGMENT
// ORDER so each lane loads its whole v16h fragment with one vector LDS read.
// Fast path (full 128-row tile, full 64-K chunk): unguarded float4 staging
// -> global_load_b128. Block: 256 threads = 8 waves.
// ---------------------------------------------------------------------------
template <int NT>
__global__ __launch_bounds__(256)
void wmma_gemm(const float* __restrict__ X, int ldx,
               const float* __restrict__ Wt, int ldw,
               const float* __restrict__ bias,
               const float* __restrict__ resid, int ldr,
               const float* __restrict__ skipg,
               float* __restrict__ Y, int ldy,
               int M, int N, int K, int act)
{
    // A fragments: [strip][kstep][half-group][row][16 halves]  (16 KB)
    __shared__ __align__(32) _Float16 sA[8][2][2][16][16];
    // B fragments: [kstep][half-group][col][16 halves]         (16 KB max)
    __shared__ __align__(32) _Float16 sB[2][2][128][16];

    const int tid  = threadIdx.x;
    const int lane = tid & 31;
    const int wave = tid >> 5;
    const int row0 = blockIdx.x * 128;
    const int hg   = lane >> 4;      // half-wave group
    const int l16  = lane & 15;
    const bool fullM = (row0 + 128 <= M);

    v8f acc[NT];
    #pragma unroll
    for (int t = 0; t < NT; ++t)
        #pragma unroll
        for (int j = 0; j < 8; ++j) acc[t][j] = 0.0f;

    const int nchunks = (K + 63) >> 6;
    for (int c = 0; c < nchunks; ++c) {
        const int k0 = c * 64;
        const bool fullK = (k0 + 64 <= K);

        if (fullM && fullK) {
            // ---------- fast path: unguarded, vectorized global reads ----------
            for (int idx = tid; idx < 512; idx += 256) {
                int r     = idx & 15;
                int g     = (idx >> 4) & 1;
                int ks    = (idx >> 5) & 1;
                int strip = idx >> 6;
                int grow  = row0 + strip * 16 + r;
                const float* xp = X + (size_t)grow * ldx + k0 + ks * 32 + g * 8;
                float4 a0 = *(const float4*)(xp);
                float4 a1 = *(const float4*)(xp + 4);
                float4 b0 = *(const float4*)(xp + 16);
                float4 b1 = *(const float4*)(xp + 20);
                v16h frag;
                frag[0]  = (_Float16)a0.x; frag[1]  = (_Float16)a0.y;
                frag[2]  = (_Float16)a0.z; frag[3]  = (_Float16)a0.w;
                frag[4]  = (_Float16)a1.x; frag[5]  = (_Float16)a1.y;
                frag[6]  = (_Float16)a1.z; frag[7]  = (_Float16)a1.w;
                frag[8]  = (_Float16)b0.x; frag[9]  = (_Float16)b0.y;
                frag[10] = (_Float16)b0.z; frag[11] = (_Float16)b0.w;
                frag[12] = (_Float16)b1.x; frag[13] = (_Float16)b1.y;
                frag[14] = (_Float16)b1.z; frag[15] = (_Float16)b1.w;
                *(v16h*)&sA[strip][ks][g][r][0] = frag;
            }
            for (int idx = tid; idx < 4 * N; idx += 256) {
                int col = idx % N;
                int q   = idx / N;
                int g   = q & 1;
                int ks  = q >> 1;
                const float* wp = Wt + (size_t)(k0 + ks * 32 + g * 8) * ldw + col;
                v16h frag;
                #pragma unroll
                for (int i = 0; i < 16; ++i)
                    frag[i] = (_Float16)wp[(size_t)(i + (i & 8)) * ldw];
                *(v16h*)&sB[ks][g][col][0] = frag;
            }
        } else {
            // ---------- guarded path (edge tiles / K not multiple of 64) ----------
            for (int idx = tid; idx < 512; idx += 256) {
                int r     = idx & 15;
                int g     = (idx >> 4) & 1;
                int ks    = (idx >> 5) & 1;
                int strip = idx >> 6;
                int grow  = row0 + strip * 16 + r;
                int kb    = k0 + ks * 32 + g * 8;
                v16h frag;
                #pragma unroll
                for (int i = 0; i < 8; ++i) {
                    int gk  = kb + i;
                    int gk2 = kb + 16 + i;
                    float v1 = (grow < M && gk  < K) ? X[(size_t)grow * ldx + gk ] : 0.0f;
                    float v2 = (grow < M && gk2 < K) ? X[(size_t)grow * ldx + gk2] : 0.0f;
                    frag[i]     = (_Float16)v1;
                    frag[8 + i] = (_Float16)v2;
                }
                *(v16h*)&sA[strip][ks][g][r][0] = frag;
            }
            for (int idx = tid; idx < 4 * N; idx += 256) {
                int col = idx % N;
                int q   = idx / N;
                int g   = q & 1;
                int ks  = q >> 1;
                int kb  = k0 + ks * 32 + g * 8;
                v16h frag;
                #pragma unroll
                for (int i = 0; i < 16; ++i) {
                    int gk = kb + i + (i & 8);
                    frag[i] = (_Float16)((gk < K) ? Wt[(size_t)gk * ldw + col] : 0.0f);
                }
                *(v16h*)&sB[ks][g][col][0] = frag;
            }
        }
        if (c + 1 < nchunks) {
            int pr = row0 + (tid >> 1);
            if (pr < M) __builtin_prefetch(&X[(size_t)pr * ldx + k0 + 64], 0, 1);
        }
        __syncthreads();

        #pragma unroll
        for (int ks = 0; ks < 2; ++ks) {
            v16h a = *(const v16h*)&sA[wave][ks][hg][l16][0];
            #pragma unroll
            for (int nt = 0; nt < NT; ++nt) {
                v16h b = *(const v16h*)&sB[ks][hg][nt * 16 + l16][0];
                acc[nt] = __builtin_amdgcn_wmma_f32_16x16x32_f16(
                    false, a, false, b, (short)0, acc[nt], false, false);
            }
        }
        __syncthreads();
    }

    // epilogue
    float sk = 0.0f, omk = 0.0f;
    const bool useSkip = (skipg != nullptr);
    if (useSkip) { float s = *skipg; sk = 1.0f / (1.0f + expf(-s)); omk = 1.0f - sk; }
    const int rbase = row0 + wave * 16 + (hg ? 8 : 0);
    #pragma unroll
    for (int nt = 0; nt < NT; ++nt) {
        int col = nt * 16 + l16;
        float bv = bias ? bias[col] : 0.0f;
        #pragma unroll
        for (int j = 0; j < 8; ++j) {
            int row = rbase + j;
            if (!fullM && row >= M) continue;
            float v = acc[nt][j] + bv;
            if (act == 1)      v = fmaxf(v, 0.0f);
            else if (act == 2) v = 0.5f * v * (1.0f + erff(v * 0.70710678118654752f));
            if (useSkip) v = sk * v + omk * resid[(size_t)row * ldr + col];
            Y[(size_t)row * ldy + col] = v;
        }
    }
}

// ---------------------------------------------------------------------------
// Graph / elementwise kernels
// ---------------------------------------------------------------------------
__device__ inline void atomicMaxF(float* addr, float val) {
    int* ia = (int*)addr;
    int old = __float_as_int(*addr);
    while (__int_as_float(old) < val) {
        int assumed = old;
        old = atomicCAS(ia, assumed, __float_as_int(val));
        if (old == assumed) break;
    }
}

__global__ void fill_val(float* p, float v, int n) {
    int i = blockIdx.x * blockDim.x + threadIdx.x;
    if (i < n) p[i] = v;
}

__global__ void edge_score(const int* __restrict__ src, const int* __restrict__ dst, int E,
                           const float* __restrict__ q, const float* __restrict__ kr,
                           const float* __restrict__ prel,
                           float* __restrict__ score, float* __restrict__ nmax)
{
    int e = blockIdx.x * blockDim.x + threadIdx.x;
    if (e >= E) return;
    int s = src[e], d = dst[e];
    #pragma unroll
    for (int h = 0; h < 2; ++h) {
        const float* qp = q  + (size_t)d * 128 + h * 64;
        const float* kp = kr + (size_t)s * 128 + h * 64;
        float acc = 0.0f;
        for (int i = 0; i < 64; i += 4) {
            float4 qa = *(const float4*)(qp + i);
            float4 ka = *(const float4*)(kp + i);
            acc += qa.x * ka.x + qa.y * ka.y + qa.z * ka.z + qa.w * ka.w;
        }
        float a = acc * prel[h] * 0.125f;   // / sqrt(64)
        score[(size_t)e * 2 + h] = a;
        atomicMaxF(&nmax[(size_t)d * 2 + h], a);
    }
}

__global__ void edge_exp(const int* __restrict__ dst, int E,
                         float* __restrict__ score, const float* __restrict__ nmax,
                         float* __restrict__ nsum)
{
    int e = blockIdx.x * blockDim.x + threadIdx.x;
    if (e >= E) return;
    int d = dst[e];
    #pragma unroll
    for (int h = 0; h < 2; ++h) {
        float m = nmax[(size_t)d * 2 + h];
        if (isinf(m) || m != m) m = 0.0f;
        float ev = expf(score[(size_t)e * 2 + h] - m);
        score[(size_t)e * 2 + h] = ev;
        atomicAdd(&nsum[(size_t)d * 2 + h], ev);
    }
}

__global__ void edge_agg(const int* __restrict__ src, const int* __restrict__ dst, int E,
                         const float* __restrict__ score, const float* __restrict__ nsum,
                         const float* __restrict__ vr, float* __restrict__ agg)
{
    int idx = blockIdx.x * blockDim.x + threadIdx.x;
    int e = idx >> 1, h = idx & 1;
    if (e >= E) return;
    int s = src[e], d = dst[e];
    float w = score[(size_t)e * 2 + h] / (nsum[(size_t)d * 2 + h] + 1e-16f);
    const float* vp = vr + (size_t)s * 128 + h * 64;
    float* ap       = agg + (size_t)d * 128 + h * 64;
    for (int i = 0; i < 64; ++i) atomicAdd(ap + i, w * vp[i]);
}

__global__ void gelu_k(float* p, int n) {
    int i = blockIdx.x * blockDim.x + threadIdx.x;
    if (i >= n) return;
    float v = p[i];
    p[i] = 0.5f * v * (1.0f + erff(v * 0.70710678118654752f));
}

__global__ void pool_accum(const float* __restrict__ x, const int* __restrict__ batch, int n,
                           float* __restrict__ pool, float* __restrict__ cnt,
                           int colOff, int cntOff)
{
    int i = blockIdx.x * blockDim.x + threadIdx.x;
    if (i >= n) return;
    int b = batch[i];
    atomicAdd(&cnt[b * 3 + cntOff], 1.0f);
    const float* xp = x + (size_t)i * 128;
    float* pp = pool + (size_t)b * 384 + colOff;
    for (int c = 0; c < 128; ++c) atomicAdd(pp + c, xp[c]);
}

__global__ void pool_fin(float* pool, const float* cnt) {
    int i = blockIdx.x * blockDim.x + threadIdx.x;
    if (i >= 64 * 384) return;
    int g = i / 384, c = i - g * 384, t = c >> 7;
    pool[i] /= fmaxf(cnt[g * 3 + t], 1.0f);
}

__global__ void mlp_out(const float* __restrict__ hbuf, const float* __restrict__ w,
                        const float* __restrict__ b, float* __restrict__ out)
{
    int g = blockIdx.x * blockDim.x + threadIdx.x;
    if (g >= 64) return;
    float acc = b[0];
    const float* hp = hbuf + (size_t)g * 128;
    for (int i = 0; i < 128; ++i) acc += hp[i] * w[i];
    out[g] = acc;
}

// ---------------------------------------------------------------------------
// Host side
// ---------------------------------------------------------------------------
static void gemm_launch(hipStream_t stream,
                        const float* X, int ldx, const float* Wt, int ldw,
                        const float* bias, const float* resid, int ldr,
                        const float* skipg, float* Y, int ldy,
                        int M, int N, int K, int act)
{
    dim3 grid((M + 127) / 128);
    if (N == 64)
        wmma_gemm<4><<<grid, 256, 0, stream>>>(X, ldx, Wt, ldw, bias, resid, ldr, skipg, Y, ldy, M, N, K, act);
    else
        wmma_gemm<8><<<grid, 256, 0, stream>>>(X, ldx, Wt, ldw, bias, resid, ldr, skipg, Y, ldy, M, N, K, act);
}

extern "C" void kernel_launch(void* const* d_in, const int* in_sizes, int n_in,
                              void* d_out, int out_size, void* d_ws, size_t ws_size,
                              hipStream_t stream) {
    // ---- flattened input map (jax tree flatten: sorted dict keys) ----
    // 0:batch_dst 1:batch_src 2:batch_units
    // 3..8: edges sorted: dst_contains_units, dst_correspond_src,
    //        src_contains_units, src_rcorrespond_dst, units_in_src, units_moveto_dst
    // 9..14: in_lin (dst.b,dst.w, src.b,src.w, units.b,units.w)
    // 15..194: 4 layers x 45: a_lin(6) a_rel(6) k(6) m_rel(6) p_rel(6) q(6) skip(3) v(6)
    // 195..198: mlp l1.b,l1.w,l2.b,l2.w ; 199..204: out_lin ; 205:x_dst 206:x_src 207:x_units
    auto F = [&](int i) { return (const float*)d_in[i]; };
    auto I = [&](int i) { return (const int*)d_in[i]; };

    // type index: 0=units 1=src_servers 2=dst_servers
    int nN[3]   = { in_sizes[2], in_sizes[1], in_sizes[0] };
    const int typeSlot[3] = { 2, 1, 0 };   // position in sorted (dst,src,units)
    const int inD[3]  = { 64, 32, 32 };
    const int xIdx[3] = { 207, 206, 205 };
    const int bIdx[3] = { 2, 1, 0 };

    const int relSrc[6] = { 2, 2, 1, 1, 0, 0 };
    const int relDst[6] = { 0, 1, 0, 2, 1, 2 };
    int relE[6];
    for (int r = 0; r < 6; ++r) relE[r] = in_sizes[3 + r] / 2;

    // ---- workspace carve-up ----
    float* Wb = (float*)d_ws;
    size_t off = 0;
    auto alloc = [&](size_t n) { float* p = Wb + off; off += n; return p; };
    float *h[3], *kb[3], *qb[3], *vb[3], *agg[3], *nmax[3], *nsum[3];
    for (int t = 0; t < 3; ++t) h[t]   = alloc((size_t)nN[t] * 128);
    for (int t = 0; t < 3; ++t) kb[t]  = alloc((size_t)nN[t] * 128);
    for (int t = 0; t < 3; ++t) qb[t]  = alloc((size_t)nN[t] * 128);
    for (int t = 0; t < 3; ++t) vb[t]  = alloc((size_t)nN[t] * 128);
    float* tmp = alloc((size_t)nN[0] * 128);          // per-relation K_r / V_r
    float* sc[6];
    for (int r = 0; r < 6; ++r) sc[r]  = alloc((size_t)relE[r] * 2);
    for (int t = 0; t < 3; ++t) nmax[t] = alloc((size_t)nN[t] * 2);
    for (int t = 0; t < 3; ++t) nsum[t] = alloc((size_t)nN[t] * 2);
    for (int t = 0; t < 3; ++t) agg[t]  = alloc((size_t)nN[t] * 128);
    float* pool = alloc(64 * 384);
    float* cnt  = alloc(64 * 3);
    float* mlph = alloc(64 * 128);
    (void)ws_size; (void)n_in; (void)out_size;

    // ---- input linears + relu ----
    for (int t = 0; t < 3; ++t) {
        int s = typeSlot[t];
        gemm_launch(stream, F(xIdx[t]), inD[t], F(9 + s * 2 + 1), 128, F(9 + s * 2),
                    nullptr, 0, nullptr, h[t], 128, nN[t], 128, inD[t], /*relu*/1);
    }

    // ---- HGT layers ----
    for (int L = 0; L < 4; ++L) {
        int base = 15 + L * 45;
        // K/Q/V projections
        for (int t = 0; t < 3; ++t) {
            int s = typeSlot[t];
            gemm_launch(stream, h[t], 128, F(base + 12 + s * 2 + 1), 128, F(base + 12 + s * 2),
                        nullptr, 0, nullptr, kb[t], 128, nN[t], 128, 128, 0);
            gemm_launch(stream, h[t], 128, F(base + 30 + s * 2 + 1), 128, F(base + 30 + s * 2),
                        nullptr, 0, nullptr, qb[t], 128, nN[t], 128, 128, 0);
            gemm_launch(stream, h[t], 128, F(base + 39 + s * 2 + 1), 128, F(base + 39 + s * 2),
                        nullptr, 0, nullptr, vb[t], 128, nN[t], 128, 128, 0);
        }
        // reset per-node softmax state + aggregates
        for (int t = 0; t < 3; ++t) {
            int n2 = nN[t] * 2;
            fill_val<<<(n2 + 255) / 256, 256, 0, stream>>>(nmax[t], -INFINITY, n2);
            hipMemsetAsync(nsum[t], 0, (size_t)n2 * sizeof(float), stream);
            hipMemsetAsync(agg[t], 0, (size_t)nN[t] * 128 * sizeof(float), stream);
        }
        // pass 1: K_r = K @ a_rel (per head), scores + segment max
        for (int r = 0; r < 6; ++r) {
            int st = relSrc[r], dt = relDst[r], E = relE[r];
            const int* ed = I(3 + r);
            for (int hh = 0; hh < 2; ++hh)
                gemm_launch(stream, kb[st] + hh * 64, 128, F(base + 6 + r) + hh * 4096, 64,
                            nullptr, nullptr, 0, nullptr, tmp + hh * 64, 128, nN[st], 64, 64, 0);
            edge_score<<<(E + 255) / 256, 256, 0, stream>>>(ed, ed + E, E, qb[dt], tmp,
                                                            F(base + 24 + r), sc[r], nmax[dt]);
        }
        // pass 2: exp + segment sum
        for (int r = 0; r < 6; ++r) {
            int dt = relDst[r], E = relE[r];
            const int* ed = I(3 + r);
            edge_exp<<<(E + 255) / 256, 256, 0, stream>>>(ed + E, E, sc[r], nmax[dt], nsum[dt]);
        }
        // pass 3: V_r = V @ m_rel, weighted scatter aggregate
        for (int r = 0; r < 6; ++r) {
            int st = relSrc[r], dt = relDst[r], E = relE[r];
            const int* ed = I(3 + r);
            for (int hh = 0; hh < 2; ++hh)
                gemm_launch(stream, vb[st] + hh * 64, 128, F(base + 18 + r) + hh * 4096, 64,
                            nullptr, nullptr, 0, nullptr, tmp + hh * 64, 128, nN[st], 64, 64, 0);
            edge_agg<<<(2 * E + 255) / 256, 256, 0, stream>>>(ed, ed + E, E, sc[r], nsum[dt], tmp, agg[dt]);
        }
        // gelu(agg) -> a_lin -> sigmoid-skip blend with h (in place)
        for (int t = 0; t < 3; ++t) {
            int s = typeSlot[t];
            int n = nN[t] * 128;
            gelu_k<<<(n + 255) / 256, 256, 0, stream>>>(agg[t], n);
            gemm_launch(stream, agg[t], 128, F(base + 0 + s * 2 + 1), 128, F(base + 0 + s * 2),
                        h[t], 128, F(base + 36 + s), h[t], 128, nN[t], 128, 128, 0);
        }
    }

    // ---- output linears (into kb as scratch) ----
    for (int t = 0; t < 3; ++t) {
        int s = typeSlot[t];
        gemm_launch(stream, h[t], 128, F(199 + s * 2 + 1), 128, F(199 + s * 2),
                    nullptr, 0, nullptr, kb[t], 128, nN[t], 128, 128, 0);
    }

    // ---- mean pool + MLP head ----
    hipMemsetAsync(pool, 0, 64 * 384 * sizeof(float), stream);
    hipMemsetAsync(cnt, 0, 64 * 3 * sizeof(float), stream);
    for (int t = 0; t < 3; ++t)
        pool_accum<<<(nN[t] + 255) / 256, 256, 0, stream>>>(kb[t], I(bIdx[t]), nN[t],
                                                            pool, cnt, t * 128, t);
    pool_fin<<<(64 * 384 + 255) / 256, 256, 0, stream>>>(pool, cnt);
    gemm_launch(stream, pool, 384, F(196), 128, F(195),
                nullptr, 0, nullptr, mlph, 128, 64, 128, 384, /*relu*/1);
    mlp_out<<<1, 64, 0, stream>>>(mlph, F(198), F(197), (float*)d_out);
}
